// QRNNLayer_72688026517799
// MI455X (gfx1250) — matrix operand order repaired and checked by reading, not compiled, not run on hardware
//
#include <hip/hip_runtime.h>
#include <hip/hip_bf16.h>

// ---------------- problem constants ----------------
#define SEQ     2048
#define NBATCH  32
#define DIM     1024
#define TD      16                 // d-columns per workgroup
#define NT      48                 // 3 gates * TD  (GEMM N)
#define SC      128                // seq rows per chunk (GEMM M)
#define NCHUNK  (SEQ / SC)         // 16
#define KTILE   32                 // wmma K per step (bf16)
#define NKT     (DIM / KTILE)      // 32
#define BLOCK   256                // 8 wave32
#define SWS     (DIM + 8)          // sW row stride (ushort): 2064 B, 16B aligned
#define SACT    50                 // sAct row stride (float), pad vs bank conflicts
#define NSEG    16                 // scan segments per chunk
#define SEGROWS (SC / NSEG)        // 8 rows per segment

#define XB_ELEMS  ((size_t)SEQ * NBATCH * DIM)          // 67,108,864
#define XB_BYTES  (XB_ELEMS * 2)                        // 128 MiB bf16 X

#define SMEM_BYTES (NT*SWS*2 + (SC*SACT + 3*NSEG*TD + TD + NT)*4)  // = 128000 B

typedef __attribute__((ext_vector_type(16))) __bf16       v16bf;
typedef __attribute__((ext_vector_type(2)))  __bf16       v2bf;
typedef __attribute__((ext_vector_type(8)))  float        v8f;
typedef __attribute__((ext_vector_type(4)))  unsigned int u32x4;
typedef __attribute__((ext_vector_type(4)))  float        f32x4;

union BF16Frag { v16bf v; u32x4 q[2]; unsigned int u[8]; };

// fp32 -> bf16 native convert (v_cvt_pk_bf16_f32); lo in [15:0], hi in [31:16]
__device__ __forceinline__ unsigned int pack_bf16(float lo, float hi) {
  v2bf p;
  p[0] = (__bf16)lo;
  p[1] = (__bf16)hi;
  return __builtin_bit_cast(unsigned int, p);
}
__device__ __forceinline__ unsigned short cvt_bf16(float f) {
  __bf16 h = (__bf16)f;
  return __builtin_bit_cast(unsigned short, h);
}
// fast sigmoid: v_exp_f32 + v_rcp_f32
__device__ __forceinline__ float sigmoidf_(float x) {
  return __builtin_amdgcn_rcpf(1.0f + __expf(-x));
}

// ---------------- pass 1: X fp32 -> bf16 (one conversion per element) --------
__global__ void __launch_bounds__(BLOCK)
x_to_bf16(const float* __restrict__ X, unsigned short* __restrict__ Xb) {
  size_t i = (size_t)blockIdx.x * BLOCK + threadIdx.x;   // 8 elements per thread
  const f32x4* src = (const f32x4*)X;
  u32x4* dst = (u32x4*)Xb;
  f32x4 x0 = src[2*i];
  f32x4 x1 = src[2*i + 1];
  u32x4 p;
  p.x = pack_bf16(x0.x, x0.y);
  p.y = pack_bf16(x0.z, x0.w);
  p.z = pack_bf16(x1.x, x1.y);
  p.w = pack_bf16(x1.z, x1.w);
  dst[i] = p;
}

// ---------------- pass 2: fused GEMM + activations + scan --------------------
template <bool PRE>
__global__ void __launch_bounds__(BLOCK)
qrnn_fused(const float* __restrict__ X, const unsigned short* __restrict__ Xb,
           const float* __restrict__ hid, const float* __restrict__ W,
           const float* __restrict__ bias, float* __restrict__ out) {
  extern __shared__ char smem[];
  unsigned short* sW   = (unsigned short*)smem;              // [NT][SWS] bf16 W tile
  float*          sAct = (float*)(smem + NT*SWS*2);          // [SC][SACT] activations
  float*          segP = sAct + SC*SACT;                     // [NSEG][TD]
  float*          segA = segP + NSEG*TD;                     // [NSEG][TD]
  float*          hseg = segA + NSEG*TD;                     // [NSEG][TD]
  float*          hcar = hseg + NSEG*TD;                     // [TD]
  float*          sBias= hcar + TD;                          // [NT]

  const int t    = threadIdx.x;
  const int lane = t & 31;
  const int wv   = t >> 5;                 // wave id 0..7 -> owns M rows wv*16..+15
  const int b    = blockIdx.x >> 6;        // DIM/TD = 64 d-tiles per batch
  const int d0   = (blockIdx.x & 63) * TD;

  // ---- stage W tile as bf16 in LDS, gate-interleaved rows: row = dd*3 + g ----
  for (int i = t; i < NT*DIM; i += BLOCK) {
    int n  = i >> 10;                      // DIM == 1024
    int k  = i & (DIM - 1);
    int dd = n / 3;
    int g  = n - 3*dd;
    sW[n*SWS + k] = cvt_bf16(W[(size_t)(g*DIM + d0 + dd)*DIM + k]);
  }
  if (t < NT) sBias[t] = bias[(t >> 4)*DIM + d0 + (t & 15)];
  if (t < TD) hcar[t]  = hid[b*DIM + d0 + t];
  __syncthreads();

  const int mrow = lane & 15;              // row within 16-row A tile / col within B tile
  const int hi   = lane >> 4;              // lane half (ISA K-split)
  const int dseg = t >> 4;                 // scan segment 0..15
  const int dcol = t & 15;                 // scan d-column 0..15

  // chunk-invariant per-lane bias
  const float bz = sBias[ 0 + mrow];
  const float bf = sBias[16 + mrow];
  const float bo = sBias[32 + mrow];

  // single B base pointer: all k-loop DS accesses are base + small immediate
  const unsigned short* bcol = sW + (mrow*3)*SWS + hi*8;

  for (int sc = 0; sc < NCHUNK; ++sc) {
    const int s0 = sc * SC;

    // ---------------- GEMM chunk: [128 x 48] += X[128 x 1024] * W^T ----------
    v8f c0 = {0,0,0,0,0,0,0,0};
    v8f c1 = {0,0,0,0,0,0,0,0};
    v8f c2 = {0,0,0,0,0,0,0,0};
    const size_t rowoff = ((size_t)(s0 + wv*16 + mrow) * NBATCH + b) * DIM + hi*8;
    const float*          xrow  = X  + rowoff;   // !PRE path
    const unsigned short* xrowb = Xb + rowoff;   //  PRE path

    #pragma unroll 8
    for (int kt = 0; kt < NKT; ++kt) {
      // A fragment: lane holds K = {kb..kb+7, kb+16..kb+23}, kb = kt*32 + hi*8
      BF16Frag a;
      if constexpr (PRE) {
        const unsigned short* xp = xrowb + kt*KTILE;
        a.q[0] = *(const u32x4*)(xp);           // K kb..kb+7   (16 B)
        a.q[1] = *(const u32x4*)(xp + 16);      // K kb+16..+23 (16 B)
      } else {
        const float* xp = xrow + kt*KTILE;
        f32x4 x0 = *(const f32x4*)(xp);
        f32x4 x1 = *(const f32x4*)(xp + 4);
        f32x4 x2 = *(const f32x4*)(xp + 16);
        f32x4 x3 = *(const f32x4*)(xp + 20);
        a.u[0] = pack_bf16(x0.x, x0.y);  a.u[1] = pack_bf16(x0.z, x0.w);
        a.u[2] = pack_bf16(x1.x, x1.y);  a.u[3] = pack_bf16(x1.z, x1.w);
        a.u[4] = pack_bf16(x2.x, x2.y);  a.u[5] = pack_bf16(x2.z, x2.w);
        a.u[6] = pack_bf16(x3.x, x3.y);  a.u[7] = pack_bf16(x3.z, x3.w);
      }

      // B fragments: lane holds col n, K = {kb..kb+7, kb+16..kb+23} (hi*8 in base)
      const unsigned short* bp = bcol + kt*KTILE;
      BF16Frag b0, b1, b2;
      b0.q[0] = *(const u32x4*)(bp        );  b0.q[1] = *(const u32x4*)(bp         + 16);
      b1.q[0] = *(const u32x4*)(bp +   SWS);  b1.q[1] = *(const u32x4*)(bp +   SWS + 16);
      b2.q[0] = *(const u32x4*)(bp + 2*SWS);  b2.q[1] = *(const u32x4*)(bp + 2*SWS + 16);

      c0 = __builtin_amdgcn_wmma_f32_16x16x32_bf16(false, a.v, false, b0.v,
                                                   (short)0, c0, false, false);
      c1 = __builtin_amdgcn_wmma_f32_16x16x32_bf16(false, a.v, false, b1.v,
                                                   (short)0, c1, false, false);
      c2 = __builtin_amdgcn_wmma_f32_16x16x32_bf16(false, a.v, false, b2.v,
                                                   (short)0, c2, false, false);
    }

    // ---------------- epilogue: bias + activations -> LDS --------------------
    {
      const int mbase = wv*16 + 8*hi;     // C layout: VGPR r, lane half -> M = r + 8*hi
      #pragma unroll
      for (int r = 0; r < 8; ++r) {
        int m = mbase + r;
        float z = c0[r] + bz;
        float f = c1[r] + bf;
        float o = c2[r] + bo;
        sAct[m*SACT +  0 + mrow] = 2.0f * sigmoidf_(2.0f * z) - 1.0f;  // tanh(z)
        sAct[m*SACT + 16 + mrow] = sigmoidf_(f);
        sAct[m*SACT + 32 + mrow] = sigmoidf_(o);
      }
    }
    __syncthreads();

    // ---------------- scan phase 1: per-segment affine reduction -------------
    {
      float P = 1.0f, A = 0.0f;
      const int mb = dseg * SEGROWS;
      #pragma unroll
      for (int i = 0; i < SEGROWS; ++i) {
        const float* row = sAct + (mb + i)*SACT;
        float z = row[dcol];
        float f = row[16 + dcol];
        A = f*A + (1.0f - f)*z;
        P *= f;
      }
      segP[dseg*TD + dcol] = P;
      segA[dseg*TD + dcol] = A;
    }
    __syncthreads();

    // ---------------- scan phase 2: segment prefix (16 threads) --------------
    if (t < TD) {
      float h = hcar[t];
      #pragma unroll
      for (int s2 = 0; s2 < NSEG; ++s2) {
        hseg[s2*TD + t] = h;
        h = segP[s2*TD + t]*h + segA[s2*TD + t];
      }
      hcar[t] = h;
    }
    __syncthreads();

    // ---------------- scan phase 3: expand + output gate + store -------------
    {
      float h = hseg[dseg*TD + dcol];
      const int mb = dseg * SEGROWS;
      size_t obase = ((size_t)(s0 + mb) * NBATCH + b) * DIM + d0 + dcol;
      #pragma unroll
      for (int i = 0; i < SEGROWS; ++i) {
        const float* row = sAct + (mb + i)*SACT;
        float z  = row[dcol];
        float f  = row[16 + dcol];
        float so = row[32 + dcol];
        h = f*h + (1.0f - f)*z;
        out[obase + (size_t)i * (NBATCH*DIM)] = so * h;
      }
    }
    __syncthreads();
  }

  // h_n = C[-1]
  if (t < TD) out[(size_t)SEQ*NBATCH*DIM + b*DIM + d0 + t] = hcar[t];
}

extern "C" void kernel_launch(void* const* d_in, const int* in_sizes, int n_in,
                              void* d_out, int out_size, void* d_ws, size_t ws_size,
                              hipStream_t stream) {
  (void)in_sizes; (void)n_in; (void)out_size;
  const float* X    = (const float*)d_in[0];
  const float* hid  = (const float*)d_in[1];
  const float* W    = (const float*)d_in[2];
  const float* bias = (const float*)d_in[3];
  float* out = (float*)d_out;

  dim3 grid(NBATCH * (DIM / TD));   // 32 batches x 64 d-tiles = 2048 workgroups

  if (d_ws != nullptr && ws_size >= XB_BYTES) {
    // pass 1: convert X once (8 elements / thread)
    unsigned short* Xb = (unsigned short*)d_ws;
    dim3 cgrid((unsigned)(XB_ELEMS / 8 / BLOCK));          // 32768 blocks
    x_to_bf16<<<cgrid, BLOCK, 0, stream>>>(X, Xb);

    hipFuncSetAttribute(reinterpret_cast<const void*>(qrnn_fused<true>),
                        hipFuncAttributeMaxDynamicSharedMemorySize, SMEM_BYTES);
    qrnn_fused<true><<<grid, BLOCK, SMEM_BYTES, stream>>>(X, Xb, hid, W, bias, out);
  } else {
    hipFuncSetAttribute(reinterpret_cast<const void*>(qrnn_fused<false>),
                        hipFuncAttributeMaxDynamicSharedMemorySize, SMEM_BYTES);
    qrnn_fused<false><<<grid, BLOCK, SMEM_BYTES, stream>>>(X, nullptr, hid, W, bias, out);
  }
}
